// CorefScore_4286377362077
// MI455X (gfx1250) — compile-verified, matrix operand order checked by reading, not run on hardware
//
#include <hip/hip_runtime.h>

// ---------------------------------------------------------------------------
// CDNA5 (gfx1250) coref forward: WMMA bf16 GEMM pipeline with TDM-staged
// weight tiles in LDS and M-register-blocking for B-fragment reuse.
// ---------------------------------------------------------------------------

typedef __attribute__((ext_vector_type(16))) __bf16       v16bf;
typedef __attribute__((ext_vector_type(8)))  float        v8f;
typedef __attribute__((ext_vector_type(4)))  unsigned int u32x4;
typedef __attribute__((ext_vector_type(4)))  int          i32x4;
typedef __attribute__((ext_vector_type(8)))  int          i32x8;

union Frag { v16bf v; u32x4 q[2]; };

#define T_LEN  2048
#define E_DIM  400
#define E_PAD  416
#define G_DIM  800
#define HS_PAD 160
#define NSPAN  20480
#define GD     1220
#define GDP    1248
#define NKEEP  384
#define NANTE  128
#define NPAIR  49152
#define PD     3680
#define KMAX   1248

// --------------------- TDM: stage a 16 x K bf16 tile into LDS --------------
// D# per ISA ch.8: group0 = {flags/count, lds_addr, global_addr, type=2},
// group1 = {data_size=2B, tensor dims/strides, tile K x 16}. Groups 2/3 zero
// (2D tile). One wave issues, waits TENSORcnt==0, then the block barriers.
__device__ __forceinline__ void tdm_load_wtile(const __bf16* gptr, unsigned lds_off,
                                               int K, int ldw) {
    unsigned long long ga = (unsigned long long)(uintptr_t)gptr;
    unsigned td0    = (unsigned)ldw;        // tensor dim0 (elements)
    unsigned td1    = 1u << 20;             // tensor dim1 (rows, generous bound)
    unsigned tile0  = (unsigned)K;          // tile dim0 = K elements
    unsigned tile1  = 16u;                  // tile dim1 = 16 rows
    unsigned long long stride = (unsigned long long)ldw;

    u32x4 g0;
    g0[0] = 1u;                                               // count=1, user D#
    g0[1] = lds_off;                                          // lds_addr (bytes)
    g0[2] = (unsigned)(ga & 0xffffffffull);                   // global_addr[31:0]
    g0[3] = (unsigned)((ga >> 32) & 0x01ffffffull) | 0x80000000u; // addr[56:32] | type=2

    i32x8 g1;
    g1[0] = 0x10000;                                          // data_size=1 (2 bytes)
    g1[1] = (int)((td0 & 0xffffu) << 16);                     // tensor_dim0[15:0]
    g1[2] = (int)(((td0 >> 16) & 0xffffu) | ((td1 & 0xffffu) << 16));
    g1[3] = (int)(((td1 >> 16) & 0xffffu) | ((tile0 & 0xffffu) << 16));
    g1[4] = (int)(tile1 & 0xffffu);                           // tile_dim1, tile_dim2=0
    g1[5] = (int)(stride & 0xffffffffull);                    // tensor_dim0_stride lo
    g1[6] = (int)((stride >> 32) & 0xffffull);                // stride hi, dim1_stride=0
    g1[7] = 0;

    i32x4 z4 = {0, 0, 0, 0};
#if defined(__clang_major__) && (__clang_major__ >= 23)
    i32x8 z8 = {0, 0, 0, 0, 0, 0, 0, 0};
    __builtin_amdgcn_tensor_load_to_lds(g0, g1, z4, z4, z8, 0);
#else
    __builtin_amdgcn_tensor_load_to_lds(g0, g1, z4, z4, 0);
#endif
    __builtin_amdgcn_s_wait_tensorcnt(0);
}

// ------------------------------- utility kernels ---------------------------

__global__ void k_fill_zero(unsigned int* p, long long n) {
    long long i = (long long)blockIdx.x * 256 + threadIdx.x;
    if (i < n) p[i] = 0u;
}

// dst (bf16, N x Kdst row-major, pre-zeroed) <- src (f32, ld src_ld) cols [col_ofs, col_ofs+K)
__global__ void k_pack_w(__bf16* dst, const float* src, int N, int K, int Kdst,
                         int src_ld, int col_ofs) {
    int i = blockIdx.x * 256 + threadIdx.x;
    if (i >= N * K) return;
    int n = i / K, k = i % K;
    dst[(long long)n * Kdst + k] = (__bf16)src[(long long)n * src_ld + col_ofs + k];
}

__global__ void k_copy_f32(float* dst, const float* src, int n) {
    int i = blockIdx.x * 256 + threadIdx.x;
    if (i < n) dst[i] = src[i];
}

__global__ void k_embed_gather(__bf16* dst, const float* emb, const int* tok) {
    int i = blockIdx.x * 256 + threadIdx.x;
    if (i >= T_LEN * E_DIM) return;
    int t = i / E_DIM, e = i % E_DIM;
    dst[(long long)t * E_PAD + e] = (__bf16)emb[(long long)tok[t] * E_DIM + e];
}

// ------------------------------- WMMA GEMM ---------------------------------
// out[M x N] = act( A[M x K]_bf16 @ W[N x K]_bf16^T + bias )
// grid = (N/16, M/(128*MB)), block = 256 (8 waves).
// The 16 x K weight tile for this block's n-tile is TDM-staged into LDS once
// and shared by all 8 waves; each wave holds MB accumulators (MB m-subtiles)
// so one B fragment feeds MB WMMAs.
template <int MB>
__global__ void k_gemm_wmma(const __bf16* __restrict__ A, const __bf16* __restrict__ W,
                            const float* __restrict__ bias,
                            float* outf, __bf16* outb,
                            int K, int lda, int ldw, int ldo, int relu) {
    __shared__ __bf16 lds_w[16 * KMAX];
    int wave = threadIdx.x >> 5;
    int lane = threadIdx.x & 31;
    int ntile = blockIdx.x * 16;
    int mbase = blockIdx.y * (128 * MB) + wave * (16 * MB);
    int kb = (lane >> 4) * 8;

    // stage the weight tile (wave 0 issues the tensor DMA)
    if (threadIdx.x < 32) {
        unsigned lds_off = (unsigned)(uintptr_t)(void*)&lds_w[0];
        tdm_load_wtile(W + (long long)ntile * ldw, lds_off, K, ldw);
    }
    __syncthreads();

    const __bf16* arow[MB];
#pragma unroll
    for (int t = 0; t < MB; ++t)
        arow[t] = A + (long long)(mbase + t * 16 + (lane & 15)) * lda + kb;
    const __bf16* brow = lds_w + (lane & 15) * K + kb;

    v8f acc[MB];
#pragma unroll
    for (int t = 0; t < MB; ++t) acc[t] = (v8f){0.f, 0.f, 0.f, 0.f, 0.f, 0.f, 0.f, 0.f};

    for (int k0 = 0; k0 < K; k0 += 32) {
        Frag b;
        b.q[0] = *(const u32x4*)(brow + k0);
        b.q[1] = *(const u32x4*)(brow + k0 + 16);
#pragma unroll
        for (int t = 0; t < MB; ++t) {
            Frag a;
            a.q[0] = *(const u32x4*)(arow[t] + k0);
            a.q[1] = *(const u32x4*)(arow[t] + k0 + 16);
            __builtin_prefetch(arow[t] + k0 + 32, 0, 1);
            acc[t] = __builtin_amdgcn_wmma_f32_16x16x32_bf16(
                false, a.v, false, b.v, (short)0, acc[t], false, false);
        }
    }

    int col = ntile + (lane & 15);
    float bn = bias ? bias[col] : 0.f;
#pragma unroll
    for (int t = 0; t < MB; ++t) {
        int mrow = mbase + t * 16 + ((lane >> 4) << 3);
#pragma unroll
        for (int r = 0; r < 8; ++r) {
            float v = acc[t][r] + bn;
            if (relu) v = fmaxf(v, 0.f);
            long long o = (long long)(mrow + r) * ldo + col;
            if (outf) outf[o] = v;
            if (outb) outb[o] = (__bf16)v;
        }
    }
}

// ------------------------------- LSTM (recurrent part) ---------------------
__global__ void k_lstm_seq(const float* __restrict__ xgf, const float* __restrict__ xgb,
                           const float* __restrict__ whhf, const float* __restrict__ whhb,
                           __bf16* states) {
    __shared__ float h[200];
    __shared__ float c[200];
    __shared__ float g[800];
    int u   = threadIdx.x;           // 0..799
    int dir = blockIdx.x;            // 0 = fwd, 1 = bwd
    const float* xg  = dir ? xgb  : xgf;
    const float* whh = dir ? whhb : whhf;
    if (u < 200) { h[u] = 0.f; c[u] = 0.f; }
    __syncthreads();
    for (int s = 0; s < T_LEN; ++s) {
        int t = dir ? (T_LEN - 1 - s) : s;
        float acc = xg[(long long)t * G_DIM + u];
        const float* wr = whh + (long long)u * 200;
        for (int j = 0; j < 200; ++j) acc += wr[j] * h[j];
        g[u] = acc;
        __syncthreads();
        float hn = 0.f, cn = 0.f;
        if (u < 200) {
            float iv = 1.f / (1.f + __expf(-g[u]));
            float fv = 1.f / (1.f + __expf(-g[200 + u]));
            float gv = tanhf(g[400 + u]);
            float ov = 1.f / (1.f + __expf(-g[600 + u]));
            cn = fv * c[u] + iv * gv;
            hn = ov * tanhf(cn);
        }
        __syncthreads();
        if (u < 200) {
            h[u] = hn; c[u] = cn;
            states[(long long)t * E_PAD + dir * 200 + u] = (__bf16)hn;
        }
        __syncthreads();
    }
}

// ------------------------------- span features -----------------------------
__global__ void k_span_feat(const __bf16* __restrict__ states, const __bf16* __restrict__ embeds,
                            const float* __restrict__ attn_out,
                            const int* __restrict__ st, const int* __restrict__ en,
                            const float* __restrict__ width_emb, __bf16* g) {
    int wave = threadIdx.x >> 5;
    int lane = threadIdx.x & 31;
    int s = blockIdx.x * 8 + wave;
    if (s >= NSPAN) return;
    int ss = st[s], ee = en[s];
    int width = ee - ss + 1;
    float av[10];
    float mx = -1e30f;
#pragma unroll
    for (int w = 0; w < 10; ++w) {
        int idx = ss + w; if (idx > T_LEN - 1) idx = T_LEN - 1;
        float a = (w < width) ? attn_out[(long long)idx * 16] : -1e9f;
        av[w] = a; mx = fmaxf(mx, a);
    }
    float ssum = 0.f;
#pragma unroll
    for (int w = 0; w < 10; ++w) { av[w] = __expf(av[w] - mx); ssum += av[w]; }
    float inv = 1.f / ssum;
    __bf16* go = g + (long long)s * GDP;
    for (int e = lane; e < E_DIM; e += 32) {
        go[e]       = states[(long long)ss * E_PAD + e];
        go[400 + e] = states[(long long)ee * E_PAD + e];
        float acc = 0.f;
#pragma unroll
        for (int w = 0; w < 10; ++w) {
            int idx = ss + w; if (idx > T_LEN - 1) idx = T_LEN - 1;
            acc += av[w] * inv * (float)embeds[(long long)idx * E_PAD + e];
        }
        go[800 + e] = (__bf16)acc;
    }
    int bn = (width >= 1) + (width >= 2) + (width >= 3) + (width >= 4) +
             (width >= 8) + (width >= 16) + (width >= 32) + (width >= 64);
    for (int d = lane; d < 20; d += 32) go[1200 + d] = (__bf16)width_emb[bn * 20 + d];
}

// ------------------------------- top-k + ordering --------------------------
extern __shared__ unsigned long long sort_smem[];
__global__ void k_topk_sort(const float* __restrict__ si_out,
                            const int* __restrict__ st, const int* __restrict__ en,
                            int* keep, int* stk, int* enk, float* skv) {
    const int NSORT = 32768;
    unsigned long long* keys = sort_smem;
    int tid = threadIdx.x;
    for (int i = tid; i < NSORT; i += 1024) {
        unsigned long long k;
        if (i < NSPAN) {
            unsigned u = __float_as_uint(si_out[(long long)i * 16]);
            unsigned skey = (u & 0x80000000u) ? ~u : (u | 0x80000000u);
            k = ((unsigned long long)(~skey) << 32) | (unsigned)i;   // asc => desc si, tie: low idx
        } else {
            k = ~0ull;
        }
        keys[i] = k;
    }
    __syncthreads();
    for (int kk = 2; kk <= NSORT; kk <<= 1) {
        for (int j = kk >> 1; j > 0; j >>= 1) {
            for (int i = tid; i < NSORT; i += 1024) {
                int ixj = i ^ j;
                if (ixj > i) {
                    unsigned long long a = keys[i], b = keys[ixj];
                    bool up = ((i & kk) == 0);
                    if ((a > b) == up) { keys[i] = b; keys[ixj] = a; }
                }
            }
            __syncthreads();
        }
    }
    // phase 2: stable argsort of kept 384 by position key (pos<<24 | rank<<15 | idx)
    unsigned long long* k2 = keys + NSORT;
    for (int r = tid; r < 512; r += 1024) {
        unsigned long long kk2;
        if (r < NKEEP) {
            int idx = (int)(keys[r] & 0xffffffffull);
            unsigned long long pos = (unsigned long long)(st[idx] * (T_LEN + 1) + en[idx]);
            kk2 = (pos << 24) | ((unsigned long long)r << 15) | (unsigned long long)idx;
        } else {
            kk2 = ~0ull;
        }
        k2[r] = kk2;
    }
    __syncthreads();
    for (int kk = 2; kk <= 512; kk <<= 1) {
        for (int j = kk >> 1; j > 0; j >>= 1) {
            for (int i = tid; i < 512; i += 1024) {
                int ixj = i ^ j;
                if (ixj > i) {
                    unsigned long long a = k2[i], b = k2[ixj];
                    bool up = ((i & kk) == 0);
                    if ((a > b) == up) { k2[i] = b; k2[ixj] = a; }
                }
            }
            __syncthreads();
        }
    }
    for (int r = tid; r < NKEEP; r += 1024) {
        int idx = (int)(k2[r] & 0x7fffull);
        keep[r] = idx;
        stk[r]  = st[idx];
        enk[r]  = en[idx];
        skv[r]  = si_out[(long long)idx * 16];
    }
}

__global__ void k_gather_gk(__bf16* gk, const __bf16* g, const int* keep) {
    int i = blockIdx.x * 256 + threadIdx.x;
    if (i >= NKEEP * GDP) return;
    int r = i / GDP, k = i % GDP;
    gk[i] = g[(long long)keep[r] * GDP + k];
}

// Dd[b][n] = dist_emb[b] . pair_w1[n, 3660:3680]
__global__ void k_dd_build(float* Dd, const float* dist_emb, const float* pw1) {
    int i = blockIdx.x * 256 + threadIdx.x;
    if (i >= 9 * HS_PAD) return;
    int b = i / HS_PAD, n = i % HS_PAD;
    float s = 0.f;
    if (n < 150)
        for (int k = 0; k < 20; ++k)
            s += dist_emb[b * 20 + k] * pw1[(long long)n * PD + 3660 + k];
    Dd[i] = s;
}

// ------------------------------- pair layer 1 ------------------------------
// h1[p][n] = ReLU( (gi*gj) @ Cp^T + U[i] + V[jc] + Dd[bin] + b1 )
// A-fragment (gi*gj) built on the fly; Cp tile TDM-staged in LDS.
__global__ void k_pair_l1(const __bf16* __restrict__ gk, const __bf16* __restrict__ Cp,
                          const float* __restrict__ U, const float* __restrict__ V,
                          const float* __restrict__ Dd, const float* __restrict__ b1,
                          const int* __restrict__ stk, const int* __restrict__ enk,
                          __bf16* outb) {
    __shared__ __bf16 lds_w[16 * GDP];
    int wave = threadIdx.x >> 5;
    int lane = threadIdx.x & 31;
    int mtile = blockIdx.y * 128 + wave * 16;
    int ntile = blockIdx.x * 16;

    if (threadIdx.x < 32) {
        unsigned lds_off = (unsigned)(uintptr_t)(void*)&lds_w[0];
        tdm_load_wtile(Cp + (long long)ntile * GDP, lds_off, GDP, GDP);
    }
    __syncthreads();

    int p = mtile + (lane & 15);
    int pi = p >> 7, pa = p & 127;
    int jr = pi - 1 - pa;
    int jc = jr < 0 ? 0 : jr;
    int kb = (lane >> 4) * 8;
    const __bf16* gi = gk + (long long)pi * GDP + kb;
    const __bf16* gj = gk + (long long)jc * GDP + kb;
    int col = ntile + (lane & 15);
    const __bf16* brow = lds_w + (lane & 15) * GDP + kb;
    v8f acc = {0.f, 0.f, 0.f, 0.f, 0.f, 0.f, 0.f, 0.f};
    for (int k0 = 0; k0 < GDP; k0 += 32) {
        Frag fa, fb, fi, fj;
        fi.q[0] = *(const u32x4*)(gi + k0);
        fi.q[1] = *(const u32x4*)(gi + k0 + 16);
        fj.q[0] = *(const u32x4*)(gj + k0);
        fj.q[1] = *(const u32x4*)(gj + k0 + 16);
#pragma unroll
        for (int e = 0; e < 16; ++e)
            fa.v[e] = (__bf16)((float)fi.v[e] * (float)fj.v[e]);
        fb.q[0] = *(const u32x4*)(brow + k0);
        fb.q[1] = *(const u32x4*)(brow + k0 + 16);
        acc = __builtin_amdgcn_wmma_f32_16x16x32_bf16(
            false, fa.v, false, fb.v, (short)0, acc, false, false);
    }
    int mrow = mtile + ((lane >> 4) << 3);
#pragma unroll
    for (int r = 0; r < 8; ++r) {
        int m = mrow + r;
        int qi = m >> 7, qa = m & 127;
        int qjr = qi - 1 - qa;
        int qjc = qjr < 0 ? 0 : qjr;
        int dist = enk[qi] - stk[qjc];
        int bn = (dist >= 1) + (dist >= 2) + (dist >= 3) + (dist >= 4) +
                 (dist >= 8) + (dist >= 16) + (dist >= 32) + (dist >= 64);
        float v = acc[r] + U[qi * HS_PAD + col] + V[qjc * HS_PAD + col] +
                  Dd[bn * HS_PAD + col] + b1[col];
        v = fmaxf(v, 0.f);
        outb[(long long)m * HS_PAD + col] = (__bf16)v;
    }
}

// ------------------------------- final output ------------------------------
__global__ void k_final(const float* __restrict__ ps, const float* __restrict__ skv,
                        float* out) {
    int i = blockIdx.x * 256 + threadIdx.x;
    if (i >= NKEEP * (NANTE + 1)) return;
    int r = i / (NANTE + 1), c = i % (NANTE + 1);
    float v;
    if (c == NANTE) {
        v = 0.f;
    } else {
        int jr = r - 1 - c;
        if (jr < 0) v = -1e9f;
        else v = skv[r] + skv[jr] + ps[(long long)(r * NANTE + c) * 16];
    }
    out[i] = v;
}

// ---------------------------------------------------------------------------

extern "C" void kernel_launch(void* const* d_in, const int* in_sizes, int n_in,
                              void* d_out, int out_size, void* d_ws, size_t ws_size,
                              hipStream_t stream) {
    (void)in_sizes; (void)n_in; (void)out_size; (void)ws_size;

    const int*   token_ids = (const int*)d_in[0];
    const int*   span_st   = (const int*)d_in[1];
    const int*   span_en   = (const int*)d_in[2];
    const float* emb       = (const float*)d_in[3];
    const float* wih_f     = (const float*)d_in[4];
    const float* whh_f     = (const float*)d_in[5];
    const float* b_f       = (const float*)d_in[6];
    const float* wih_b     = (const float*)d_in[7];
    const float* whh_b     = (const float*)d_in[8];
    const float* b_b       = (const float*)d_in[9];
    const float* aw1 = (const float*)d_in[10]; const float* ab1 = (const float*)d_in[11];
    const float* aw2 = (const float*)d_in[12]; const float* ab2 = (const float*)d_in[13];
    const float* aw3 = (const float*)d_in[14]; const float* ab3 = (const float*)d_in[15];
    const float* width_emb = (const float*)d_in[16];
    const float* mw1 = (const float*)d_in[17]; const float* mb1 = (const float*)d_in[18];
    const float* mw2 = (const float*)d_in[19]; const float* mb2 = (const float*)d_in[20];
    const float* mw3 = (const float*)d_in[21]; const float* mb3 = (const float*)d_in[22];
    const float* dist_emb = (const float*)d_in[23];
    const float* pw1 = (const float*)d_in[24]; const float* pb1 = (const float*)d_in[25];
    const float* pw2 = (const float*)d_in[26]; const float* pb2 = (const float*)d_in[27];
    const float* pw3 = (const float*)d_in[28]; const float* pb3 = (const float*)d_in[29];

    // ---- workspace layout (256B aligned) ----
    char* ws = (char*)d_ws;
    size_t off = 0;
    auto alloc = [&](size_t bytes) -> char* {
        char* p = ws + off;
        off = (off + bytes + 255) & ~(size_t)255;
        return p;
    };
    __bf16* embeds_b = (__bf16*)alloc((size_t)T_LEN * E_PAD * 2);
    __bf16* wihf_b   = (__bf16*)alloc((size_t)G_DIM * E_PAD * 2);
    __bf16* wihb_b   = (__bf16*)alloc((size_t)G_DIM * E_PAD * 2);
    float*  xgf      = (float*) alloc((size_t)T_LEN * G_DIM * 4);
    float*  xgb      = (float*) alloc((size_t)T_LEN * G_DIM * 4);
    __bf16* states_b = (__bf16*)alloc((size_t)T_LEN * E_PAD * 2);
    __bf16* aw1p = (__bf16*)alloc((size_t)HS_PAD * E_PAD * 2);
    __bf16* aw2p = (__bf16*)alloc((size_t)HS_PAD * HS_PAD * 2);
    __bf16* aw3p = (__bf16*)alloc((size_t)16 * HS_PAD * 2);
    float*  ab1p = (float*)alloc(HS_PAD * 4);
    float*  ab2p = (float*)alloc(HS_PAD * 4);
    float*  ab3p = (float*)alloc(16 * 4);
    __bf16* ah1  = (__bf16*)alloc((size_t)T_LEN * HS_PAD * 2);
    __bf16* ah2  = (__bf16*)alloc((size_t)T_LEN * HS_PAD * 2);
    float*  attn_out = (float*)alloc((size_t)T_LEN * 16 * 4);
    __bf16* g_b  = (__bf16*)alloc((size_t)NSPAN * GDP * 2);
    __bf16* mw1p = (__bf16*)alloc((size_t)HS_PAD * GDP * 2);
    __bf16* mw2p = (__bf16*)alloc((size_t)HS_PAD * HS_PAD * 2);
    __bf16* mw3p = (__bf16*)alloc((size_t)16 * HS_PAD * 2);
    float*  mb1p = (float*)alloc(HS_PAD * 4);
    float*  mb2p = (float*)alloc(HS_PAD * 4);
    float*  mb3p = (float*)alloc(16 * 4);
    __bf16* mh1  = (__bf16*)alloc((size_t)NSPAN * HS_PAD * 2);
    __bf16* mh2  = (__bf16*)alloc((size_t)NSPAN * HS_PAD * 2);
    float*  si_out = (float*)alloc((size_t)NSPAN * 16 * 4);
    int*    keep = (int*)alloc(NKEEP * 4);
    int*    stk  = (int*)alloc(NKEEP * 4);
    int*    enk  = (int*)alloc(NKEEP * 4);
    float*  skv  = (float*)alloc(NKEEP * 4);
    __bf16* gk_b = (__bf16*)alloc((size_t)NKEEP * GDP * 2);
    __bf16* Ap   = (__bf16*)alloc((size_t)HS_PAD * GDP * 2);
    __bf16* Bp   = (__bf16*)alloc((size_t)HS_PAD * GDP * 2);
    __bf16* Cp   = (__bf16*)alloc((size_t)HS_PAD * GDP * 2);
    __bf16* pw2p = (__bf16*)alloc((size_t)HS_PAD * HS_PAD * 2);
    __bf16* pw3p = (__bf16*)alloc((size_t)16 * HS_PAD * 2);
    float*  pb1p = (float*)alloc(HS_PAD * 4);
    float*  pb2p = (float*)alloc(HS_PAD * 4);
    float*  pb3p = (float*)alloc(16 * 4);
    float*  Dd   = (float*)alloc((size_t)9 * HS_PAD * 4);
    float*  Uf   = (float*)alloc((size_t)NKEEP * HS_PAD * 4);
    float*  Vf   = (float*)alloc((size_t)NKEEP * HS_PAD * 4);
    __bf16* ph1  = (__bf16*)alloc((size_t)NPAIR * HS_PAD * 2);
    __bf16* ph2  = (__bf16*)alloc((size_t)NPAIR * HS_PAD * 2);
    float*  ps   = (float*)alloc((size_t)NPAIR * 16 * 4);

    // zero the workspace (padded K columns / N rows must be 0)
    long long words = (long long)(off / 4);
    k_fill_zero<<<(unsigned)((words + 255) / 256), 256, 0, stream>>>((unsigned int*)d_ws, words);

    auto pack = [&](__bf16* dst, const float* src, int N, int K, int Kd, int ld, int co) {
        int n = N * K;
        k_pack_w<<<(n + 255) / 256, 256, 0, stream>>>(dst, src, N, K, Kd, ld, co);
    };
    auto copyb = [&](float* dst, const float* src, int n) {
        k_copy_f32<<<(n + 255) / 256, 256, 0, stream>>>(dst, src, n);
    };
    // MB=2 path: M multiple of 256 (all the big GEMMs)
    auto gemm2 = [&](const __bf16* A, const __bf16* W, const float* bias,
                     float* outf, __bf16* outb, int M, int N, int K,
                     int lda, int ldw, int ldo, int relu) {
        dim3 grid(N / 16, M / 256);
        k_gemm_wmma<2><<<grid, 256, 0, stream>>>(A, W, bias, outf, outb, K, lda, ldw, ldo, relu);
    };
    // MB=1 path: small-M GEMMs (M multiple of 128)
    auto gemm1 = [&](const __bf16* A, const __bf16* W, const float* bias,
                     float* outf, __bf16* outb, int M, int N, int K,
                     int lda, int ldw, int ldo, int relu) {
        dim3 grid(N / 16, M / 128);
        k_gemm_wmma<1><<<grid, 256, 0, stream>>>(A, W, bias, outf, outb, K, lda, ldw, ldo, relu);
    };

    // ---- weight repacks (f32 -> padded bf16) ----
    pack(wihf_b, wih_f, G_DIM, E_DIM, E_PAD, E_DIM, 0);
    pack(wihb_b, wih_b, G_DIM, E_DIM, E_PAD, E_DIM, 0);
    pack(aw1p, aw1, 150, E_DIM, E_PAD, E_DIM, 0);
    pack(aw2p, aw2, 150, 150, HS_PAD, 150, 0);
    pack(aw3p, aw3, 1, 150, HS_PAD, 150, 0);
    copyb(ab1p, ab1, 150); copyb(ab2p, ab2, 150); copyb(ab3p, ab3, 1);
    pack(mw1p, mw1, 150, GD, GDP, GD, 0);
    pack(mw2p, mw2, 150, 150, HS_PAD, 150, 0);
    pack(mw3p, mw3, 1, 150, HS_PAD, 150, 0);
    copyb(mb1p, mb1, 150); copyb(mb2p, mb2, 150); copyb(mb3p, mb3, 1);
    pack(Ap, pw1, 150, GD, GDP, PD, 0);
    pack(Bp, pw1, 150, GD, GDP, PD, GD);
    pack(Cp, pw1, 150, GD, GDP, PD, 2 * GD);
    pack(pw2p, pw2, 150, 150, HS_PAD, 150, 0);
    pack(pw3p, pw3, 1, 150, HS_PAD, 150, 0);
    copyb(pb1p, pb1, 150); copyb(pb2p, pb2, 150); copyb(pb3p, pb3, 1);
    k_dd_build<<<(9 * HS_PAD + 255) / 256, 256, 0, stream>>>(Dd, dist_emb, pw1);

    // ---- embedding gather ----
    k_embed_gather<<<(T_LEN * E_DIM + 255) / 256, 256, 0, stream>>>(embeds_b, emb, token_ids);

    // ---- LSTM: input GEMMs (WMMA) then sequential recurrence ----
    gemm2(embeds_b, wihf_b, b_f, xgf, nullptr, T_LEN, G_DIM, E_PAD, E_PAD, E_PAD, G_DIM, 0);
    gemm2(embeds_b, wihb_b, b_b, xgb, nullptr, T_LEN, G_DIM, E_PAD, E_PAD, E_PAD, G_DIM, 0);
    k_lstm_seq<<<2, 800, 0, stream>>>(xgf, xgb, whh_f, whh_b, states_b);

    // ---- attention head MLP ----
    gemm2(states_b, aw1p, ab1p, nullptr, ah1, T_LEN, HS_PAD, E_PAD, E_PAD, E_PAD, HS_PAD, 1);
    gemm2(ah1, aw2p, ab2p, nullptr, ah2, T_LEN, HS_PAD, HS_PAD, HS_PAD, HS_PAD, HS_PAD, 1);
    gemm2(ah2, aw3p, ab3p, attn_out, nullptr, T_LEN, 16, HS_PAD, HS_PAD, HS_PAD, 16, 0);

    // ---- span features ----
    k_span_feat<<<NSPAN / 8, 256, 0, stream>>>(states_b, embeds_b, attn_out,
                                               span_st, span_en, width_emb, g_b);

    // ---- mention MLP ----
    gemm2(g_b, mw1p, mb1p, nullptr, mh1, NSPAN, HS_PAD, GDP, GDP, GDP, HS_PAD, 1);
    gemm2(mh1, mw2p, mb2p, nullptr, mh2, NSPAN, HS_PAD, HS_PAD, HS_PAD, HS_PAD, HS_PAD, 1);
    gemm2(mh2, mw3p, mb3p, si_out, nullptr, NSPAN, 16, HS_PAD, HS_PAD, HS_PAD, 16, 0);

    // ---- top-k + positional ordering (single-WGP bitonic in 260KB LDS) ----
    k_topk_sort<<<1, 1024, (32768 + 512) * sizeof(unsigned long long), stream>>>(
        si_out, span_st, span_en, keep, stk, enk, skv);
    k_gather_gk<<<(NKEEP * GDP + 255) / 256, 256, 0, stream>>>(gk_b, g_b, keep);

    // ---- pair stage: U = gk@A^T, V = gk@B^T (tiny WMMA GEMMs) ----
    gemm1(gk_b, Ap, nullptr, Uf, nullptr, NKEEP, HS_PAD, GDP, GDP, GDP, HS_PAD, 0);
    gemm1(gk_b, Bp, nullptr, Vf, nullptr, NKEEP, HS_PAD, GDP, GDP, GDP, HS_PAD, 0);

    // ---- pair layer 1: fused (gi*gj)@C^T + U + V + Dd + b1, ReLU ----
    {
        dim3 grid(HS_PAD / 16, NPAIR / 128);
        k_pair_l1<<<grid, 256, 0, stream>>>(gk_b, Cp, Uf, Vf, Dd, pb1p, stk, enk, ph1);
    }
    gemm2(ph1, pw2p, pb2p, nullptr, ph2, NPAIR, HS_PAD, HS_PAD, HS_PAD, HS_PAD, HS_PAD, 1);
    gemm2(ph2, pw3p, pb3p, ps, nullptr, NPAIR, 16, HS_PAD, HS_PAD, HS_PAD, 16, 0);

    // ---- final sij assembly ----
    k_final<<<(NKEEP * (NANTE + 1) + 255) / 256, 256, 0, stream>>>(ps, skv, (float*)d_out);
}
